// GNNFingerprint2D_1984274891280
// MI455X (gfx1250) — matrix operand order repaired and checked by reading, not compiled, not run on hardware
//
#include <hip/hip_runtime.h>

#define N_NODES  98304
#define N_GRAPHS 2048
#define NODES_PER_GRAPH 48
#define N_EDGES  393216
#define NODE_IN  128
#define EMB      64
#define HID      256
#define N_LAYERS 6
#define ATTN_DIM 256
#define OUT_DIM  1024

typedef __bf16 v16bf __attribute__((ext_vector_type(16)));
typedef float  v8f   __attribute__((ext_vector_type(8)));

// ---------------------------------------------------------------- WMMA helpers
__device__ __forceinline__ v8f wmma_bf16(v16bf a, v16bf b, v8f c) {
    return __builtin_amdgcn_wmma_f32_16x16x32_bf16(false, a, false, b, (short)0, c, false, false);
}

// A fragment (16x32 bf16, MxK) from row-major bf16 in LDS (stride lda elems).
// lanes 0-15: M=lane, K = k0+0..15 (elems 0-15)
// lanes 16-31: M=lane-16, K = k0+16..31
__device__ __forceinline__ v16bf lds_a_frag(const __bf16* base, int lda, int k0, int lane) {
    int m  = lane & 15;
    int kb = k0 + ((lane & 16) ? 16 : 0);
    const __bf16* p = base + m * lda + kb;
    v16bf a;
#pragma unroll
    for (int i = 0; i < 8; ++i) { a[i] = p[i]; a[i + 8] = p[i + 8]; }
    return a;
}

// B fragment from PACKED weights: each 32x16 (KxN) tile is stored as
// 32 lanes x 16 contiguous bf16 (32B per lane -> two global_load_b128).
// Packing encodes: lanes 0-15: N=lane, K=kt*32+0..15 ; lanes 16-31: N=lane-16, K=kt*32+16..31
__device__ __forceinline__ v16bf pk_b_frag(const __bf16* P, int nTiles, int kt, int nt, int lane) {
    return *(const v16bf*)(P + ((((long)kt * nTiles + nt) << 5) + lane) * 16);
}

// Store C (16x16 f32) fragment to LDS row-major bf16, optional relu.
__device__ __forceinline__ void lds_store_c(__bf16* base, int ldn, int n0, int mOff,
                                            v8f acc, int lane, bool relu) {
    int col   = n0 + (lane & 15);
    int rbase = mOff + ((lane & 16) ? 8 : 0);
#pragma unroll
    for (int r = 0; r < 8; ++r) {
        float v = acc[r];
        if (relu && v < 0.f) v = 0.f;
        base[(rbase + r) * ldn + col] = (__bf16)v;
    }
}

// ---------------------------------------------------------------- small kernels
__global__ void zero_kernel(float* p, int n) {
    int i = blockIdx.x * blockDim.x + threadIdx.x;
    if (i < n) p[i] = 0.f;
}

// Convert f32 weights [K][N] -> bf16 packed fragment-major layout.
// dst idx = ((kt*(N/16)+nt)*32 + lane)*16 + s  with  n = nt*16+(lane&15),
// k = kt*32 + ((lane&16)?16:0) + s
__global__ void pack_kernel(const float* __restrict__ W, __bf16* __restrict__ P, int K, int N) {
    int idx = blockIdx.x * blockDim.x + threadIdx.x;
    if (idx >= K * N) return;
    int tile = idx >> 9;          // 512 elems per 32x16 tile
    int r    = idx & 511;
    int lane = r >> 4, s = r & 15;
    int nTiles = N >> 4;
    int kt = tile / nTiles, nt = tile - kt * nTiles;
    int n = (nt << 4) + (lane & 15);
    int k = (kt << 5) + ((lane & 16) ? 16 : 0) + s;
    P[idx] = (__bf16)W[(long)k * N + n];
}

__global__ void deg_kernel(const int* __restrict__ row, float* __restrict__ deg) {
    int e = blockIdx.x * blockDim.x + threadIdx.x;
    if (e < N_EDGES) atomicAdd(&deg[row[e]], 1.0f);
}

// ---------------------------------------------------------------- generic WMMA GEMM
// C[M,N] = act( act_a(A[M,K]) @ Wpacked[K,N] + bias[N] ), M_TILE=64 per block, K<=256.
__global__ __launch_bounds__(256) void gemm_kernel(
    const float* __restrict__ A, const __bf16* __restrict__ W,
    const float* __restrict__ bias, float* __restrict__ C,
    int K, int N, int reluA, int reluOut) {
    __shared__ __bf16 sA[64 * 256];
    int tid = threadIdx.x, lane = tid & 31, wave = tid >> 5;
    long m0 = (long)blockIdx.x * 64;
    const float* Ab = A + m0 * K;
    for (int i = tid; i < 64 * K; i += 256) {
        float v = Ab[i];
        if (reluA && v < 0.f) v = 0.f;
        sA[i] = (__bf16)v;
    }
    __syncthreads();
    int nT = N >> 4;
    int kT = K >> 5;
    for (int t = wave; t < nT; t += 8) {
        int n0 = t << 4;
        float bv = bias[n0 + (lane & 15)];
        v8f acc[4];
#pragma unroll
        for (int mi = 0; mi < 4; ++mi)
#pragma unroll
            for (int i = 0; i < 8; ++i) acc[mi][i] = bv;
        for (int kt = 0; kt < kT; ++kt) {
            v16bf b = pk_b_frag(W, nT, kt, t, lane);
#pragma unroll
            for (int mi = 0; mi < 4; ++mi) {
                v16bf a = lds_a_frag(sA + mi * 16 * K, K, kt << 5, lane);
                acc[mi] = wmma_bf16(a, b, acc[mi]);
            }
        }
        int col = n0 + (lane & 15);
        int rb  = (lane & 16) ? 8 : 0;
#pragma unroll
        for (int mi = 0; mi < 4; ++mi)
#pragma unroll
            for (int r = 0; r < 8; ++r) {
                float v = acc[mi][r];
                if (reluOut && v < 0.f) v = 0.f;
                C[(m0 + mi * 16 + rb + r) * N + col] = v;
            }
    }
}

// ---------------------------------------------------------------- fused GNN layer
// h[n] += deg[n] * ( relu(relu(h@W1+b1)@W2+b2) @ W3 + b3 ), 32 nodes / block.
__global__ __launch_bounds__(256) void gnn_layer_kernel(
    float* __restrict__ h,
    const __bf16* __restrict__ W1, const float* __restrict__ b1,
    const __bf16* __restrict__ W2, const float* __restrict__ b2,
    const __bf16* __restrict__ W3, const float* __restrict__ b3,
    const float* __restrict__ deg) {
    __shared__ __bf16 sh[32 * 64];     //  4 KB
    __shared__ __bf16 sm1[32 * 256];   // 16 KB
    __shared__ __bf16 sm2[32 * 512];   // 32 KB
    __shared__ float  sdeg[32];
    int tid = threadIdx.x, lane = tid & 31, wave = tid >> 5;
    long m0 = (long)blockIdx.x * 32;

    for (int i = tid; i < 32 * 64; i += 256) sh[i] = (__bf16)h[m0 * 64 + i];
    if (tid < 32) sdeg[tid] = deg[m0 + tid];
    __syncthreads();

    // stage 1: m1 = relu(h @ W1 + b1)   K=64, N=256 (16 n-tiles)
    for (int t = wave; t < 16; t += 8) {
        int n0 = t << 4;
        float bv = b1[n0 + (lane & 15)];
        v8f a0, a1;
#pragma unroll
        for (int i = 0; i < 8; ++i) { a0[i] = bv; a1[i] = bv; }
#pragma unroll
        for (int kt = 0; kt < 2; ++kt) {
            v16bf b = pk_b_frag(W1, 16, kt, t, lane);
            a0 = wmma_bf16(lds_a_frag(sh, 64, kt << 5, lane), b, a0);
            a1 = wmma_bf16(lds_a_frag(sh + 16 * 64, 64, kt << 5, lane), b, a1);
        }
        lds_store_c(sm1, 256, n0, 0, a0, lane, true);
        lds_store_c(sm1, 256, n0, 16, a1, lane, true);
    }
    __syncthreads();

    // stage 2: m2 = relu(m1 @ W2 + b2)  K=256, N=512 (32 n-tiles)
    for (int t = wave; t < 32; t += 8) {
        int n0 = t << 4;
        float bv = b2[n0 + (lane & 15)];
        v8f a0, a1;
#pragma unroll
        for (int i = 0; i < 8; ++i) { a0[i] = bv; a1[i] = bv; }
#pragma unroll
        for (int kt = 0; kt < 8; ++kt) {
            v16bf b = pk_b_frag(W2, 32, kt, t, lane);
            a0 = wmma_bf16(lds_a_frag(sm1, 256, kt << 5, lane), b, a0);
            a1 = wmma_bf16(lds_a_frag(sm1 + 16 * 256, 256, kt << 5, lane), b, a1);
        }
        lds_store_c(sm2, 512, n0, 0, a0, lane, true);
        lds_store_c(sm2, 512, n0, 16, a1, lane, true);
    }
    __syncthreads();

    // stage 3: dh = m2 @ W3 + b3 ; h += deg * dh   K=512, N=64 (4 n-tiles x 2 m-tiles)
    {
        int t  = wave & 3;
        int n0 = t << 4;
        int mi = wave >> 2;           // 0 or 1
        float bv = b3[n0 + (lane & 15)];
        v8f acc;
#pragma unroll
        for (int i = 0; i < 8; ++i) acc[i] = bv;
#pragma unroll
        for (int kt = 0; kt < 16; ++kt) {
            v16bf b = pk_b_frag(W3, 4, kt, t, lane);
            acc = wmma_bf16(lds_a_frag(sm2 + mi * 16 * 512, 512, kt << 5, lane), b, acc);
        }
        int col = n0 + (lane & 15);
        int rb  = (lane & 16) ? 8 : 0;
#pragma unroll
        for (int r = 0; r < 8; ++r) {
            int ml = mi * 16 + rb + r;
            long row = m0 + ml;
            h[row * 64 + col] += sdeg[ml] * acc[r];
        }
    }
}

// ---------------------------------------------------------------- attention prep
// qh = (q@Wq+bq)/8 ; u[h][k] = sum_d Wk[k, h*64+d]*qh[h,d] ; c[h] = bk_h . qh_h
__global__ __launch_bounds__(256) void qprep_kernel(
    const float* __restrict__ q, const float* __restrict__ W_in,
    const float* __restrict__ b_in, float* __restrict__ u, float* __restrict__ c) {
    __shared__ float sqh[256];
    int tid = threadIdx.x;
    float s = b_in[tid];
    for (int d = 0; d < 256; ++d) s += q[d] * W_in[d * 768 + tid];
    sqh[tid] = s * 0.125f;
    __syncthreads();
    for (int hh = 0; hh < 4; ++hh) {
        float acc = 0.f;
        for (int d = 0; d < 64; ++d)
            acc += W_in[tid * 768 + 256 + hh * 64 + d] * sqh[hh * 64 + d];
        u[hh * 256 + tid] = acc;
    }
    if (tid < 4) {
        float acc = 0.f;
        for (int d = 0; d < 64; ++d) acc += b_in[256 + tid * 64 + d] * sqh[tid * 64 + d];
        c[tid] = acc;
    }
}

__global__ void scores_kernel(const float* __restrict__ hp, const float* __restrict__ u,
                              const float* __restrict__ c, float* __restrict__ scores) {
    int t = blockIdx.x * blockDim.x + threadIdx.x;
    if (t >= N_NODES * 4) return;
    int n = t >> 2, hh = t & 3;
    const float* row = hp + (long)n * 256;
    const float* uh  = u + hh * 256;
    float s = 0.f;
    for (int d = 0; d < 256; ++d) s += row[d] * uh[d];
    scores[t] = s + c[hh];
}

// per-graph softmax over 48 nodes, pooled = (sum_n w*hp) @ Wv + bv
__global__ __launch_bounds__(256) void pool_kernel(
    const float* __restrict__ hp, const float* __restrict__ scores,
    const float* __restrict__ W_in, const float* __restrict__ b_in,
    float* __restrict__ pooled) {
    __shared__ float sE[192];
    __shared__ float sMax[4], sSum[4];
    __shared__ float sP[4 * 256];
    int g = blockIdx.x, tid = threadIdx.x;
    long base = (long)g * NODES_PER_GRAPH;
    if (tid < 192) sE[tid] = scores[base * 4 + tid];
    __syncthreads();
    if (tid < 4) {
        float m = -1e30f;
        for (int i = 0; i < 48; ++i) { float v = sE[i * 4 + tid]; m = v > m ? v : m; }
        sMax[tid] = m;
    }
    __syncthreads();
    if (tid < 192) sE[tid] = expf(sE[tid] - sMax[tid & 3]);
    __syncthreads();
    if (tid < 4) {
        float s = 0.f;
        for (int i = 0; i < 48; ++i) s += sE[i * 4 + tid];
        sSum[tid] = s;
    }
    __syncthreads();
    {
        float p0 = 0.f, p1 = 0.f, p2 = 0.f, p3 = 0.f;
        for (int i = 0; i < 48; ++i) {
            float v = hp[(base + i) * 256 + tid];
            p0 += sE[i * 4 + 0] * v; p1 += sE[i * 4 + 1] * v;
            p2 += sE[i * 4 + 2] * v; p3 += sE[i * 4 + 3] * v;
        }
        sP[0 * 256 + tid] = p0 / sSum[0];
        sP[1 * 256 + tid] = p1 / sSum[1];
        sP[2 * 256 + tid] = p2 / sSum[2];
        sP[3 * 256 + tid] = p3 / sSum[3];
    }
    __syncthreads();
    int hh = tid >> 6;
    float acc = b_in[512 + tid];
    const float* Ph = sP + hh * 256;
    for (int k = 0; k < 256; ++k) acc += Ph[k] * W_in[k * 768 + 512 + tid];
    pooled[(long)g * 256 + tid] = acc;
}

__global__ __launch_bounds__(256) void ln_kernel(
    const float* __restrict__ X, const float* __restrict__ g,
    const float* __restrict__ b, float* __restrict__ Y) {
    __shared__ float red[256];
    int row = blockIdx.x, tid = threadIdx.x;
    float v = X[(long)row * 256 + tid];
    red[tid] = v; __syncthreads();
    for (int s = 128; s > 0; s >>= 1) { if (tid < s) red[tid] += red[tid + s]; __syncthreads(); }
    float mu = red[0] * (1.f / 256.f); __syncthreads();
    float d = v - mu;
    red[tid] = d * d; __syncthreads();
    for (int s = 128; s > 0; s >>= 1) { if (tid < s) red[tid] += red[tid + s]; __syncthreads(); }
    float var = red[0] * (1.f / 256.f);
    Y[(long)row * 256 + tid] = d * rsqrtf(var + 1e-5f) * g[tid] + b[tid];
}

// ---------------------------------------------------------------- host
extern "C" void kernel_launch(void* const* d_in, const int* in_sizes, int n_in,
                              void* d_out, int out_size, void* d_ws, size_t ws_size,
                              hipStream_t stream) {
    const float* x       = (const float*)d_in[0];
    const int*   edge    = (const int*)d_in[1];     // edge_index[0] = first N_EDGES ints
    const float* W_embed = (const float*)d_in[3];
    const float* b_embed = (const float*)d_in[4];
    const float* W1      = (const float*)d_in[5];
    const float* b1      = (const float*)d_in[6];
    const float* W2      = (const float*)d_in[7];
    const float* b2      = (const float*)d_in[8];
    const float* W3      = (const float*)d_in[9];
    const float* b3      = (const float*)d_in[10];
    const float* Wp      = (const float*)d_in[11];
    const float* bp      = (const float*)d_in[12];
    const float* q       = (const float*)d_in[13];
    const float* W_in    = (const float*)d_in[14];
    const float* b_in    = (const float*)d_in[15];
    const float* W_out   = (const float*)d_in[16];
    const float* b_out   = (const float*)d_in[17];
    const float* Wh1     = (const float*)d_in[18];
    const float* bh1     = (const float*)d_in[19];
    const float* ln_g    = (const float*)d_in[20];
    const float* ln_b    = (const float*)d_in[21];
    const float* Wh2     = (const float*)d_in[22];
    const float* bh2     = (const float*)d_in[23];

    char* ws = (char*)d_ws;
    size_t off = 0;
    auto alloc = [&](size_t bytes) -> char* {
        char* p = ws + off;
        off += (bytes + 255) & ~(size_t)255;
        return p;
    };
    float* deg    = (float*)alloc((size_t)N_NODES * 4);
    float* h      = (float*)alloc((size_t)N_NODES * EMB * 4);
    float* hp     = (float*)alloc((size_t)N_NODES * ATTN_DIM * 4);
    float* sc     = (float*)alloc((size_t)N_NODES * 4 * 4);
    float* u      = (float*)alloc(4 * 256 * 4);
    float* c      = (float*)alloc(4 * 4);
    float* pooled = (float*)alloc((size_t)N_GRAPHS * 256 * 4);
    float* attn   = (float*)alloc((size_t)N_GRAPHS * 256 * 4);
    float* t1     = (float*)alloc((size_t)N_GRAPHS * 256 * 4);
    float* t2     = (float*)alloc((size_t)N_GRAPHS * 256 * 4);
    __bf16* Wemb_b = (__bf16*)alloc((size_t)128 * 64 * 2);
    __bf16* W1b    = (__bf16*)alloc((size_t)6 * 64 * 256 * 2);
    __bf16* W2b    = (__bf16*)alloc((size_t)6 * 256 * 512 * 2);
    __bf16* W3b    = (__bf16*)alloc((size_t)6 * 512 * 64 * 2);
    __bf16* Wpb    = (__bf16*)alloc((size_t)64 * 256 * 2);
    __bf16* Woutb  = (__bf16*)alloc((size_t)256 * 256 * 2);
    __bf16* Wh1b   = (__bf16*)alloc((size_t)256 * 256 * 2);
    __bf16* Wh2b   = (__bf16*)alloc((size_t)256 * 1024 * 2);

    auto pack = [&](const float* s, __bf16* d, int K, int N) {
        pack_kernel<<<(K * N + 255) / 256, 256, 0, stream>>>(s, d, K, N);
    };
    pack(W_embed, Wemb_b, 128, 64);
    for (int l = 0; l < N_LAYERS; ++l) {
        pack(W1 + (size_t)l * 64 * 256,  W1b + (size_t)l * 64 * 256,  64, 256);
        pack(W2 + (size_t)l * 256 * 512, W2b + (size_t)l * 256 * 512, 256, 512);
        pack(W3 + (size_t)l * 512 * 64,  W3b + (size_t)l * 512 * 64,  512, 64);
    }
    pack(Wp, Wpb, 64, 256);
    pack(W_out, Woutb, 256, 256);
    pack(Wh1, Wh1b, 256, 256);
    pack(Wh2, Wh2b, 256, 1024);

    // degree histogram (edge_index[1] is never used by the reference math)
    zero_kernel<<<(N_NODES + 255) / 256, 256, 0, stream>>>(deg, N_NODES);
    deg_kernel<<<(N_EDGES + 255) / 256, 256, 0, stream>>>(edge, deg);

    // h = x @ W_embed + b_embed
    gemm_kernel<<<N_NODES / 64, 256, 0, stream>>>(x, Wemb_b, b_embed, h, 128, 64, 0, 0);

    // 6 fused layers: h += deg * MLP(h)
    for (int l = 0; l < N_LAYERS; ++l)
        gnn_layer_kernel<<<N_NODES / 32, 256, 0, stream>>>(
            h, W1b + (size_t)l * 64 * 256, b1 + (size_t)l * 256,
            W2b + (size_t)l * 256 * 512, b2 + (size_t)l * 512,
            W3b + (size_t)l * 512 * 64, b3 + (size_t)l * 64, deg);

    // hp = h @ Wp + bp
    gemm_kernel<<<N_NODES / 64, 256, 0, stream>>>(h, Wpb, bp, hp, 64, 256, 0, 0);

    // attention: folded q / scores / softmax-pool
    qprep_kernel<<<1, 256, 0, stream>>>(q, W_in, b_in, u, c);
    scores_kernel<<<(N_NODES * 4 + 255) / 256, 256, 0, stream>>>(hp, u, c, sc);
    pool_kernel<<<N_GRAPHS, 256, 0, stream>>>(hp, sc, W_in, b_in, pooled);

    // tail MLP
    gemm_kernel<<<N_GRAPHS / 64, 256, 0, stream>>>(pooled, Woutb, b_out, attn, 256, 256, 0, 0);
    gemm_kernel<<<N_GRAPHS / 64, 256, 0, stream>>>(attn, Wh1b, bh1, t1, 256, 256, 1, 1);
    ln_kernel<<<N_GRAPHS, 256, 0, stream>>>(t1, ln_g, ln_b, t2);
    gemm_kernel<<<N_GRAPHS / 64, 256, 0, stream>>>(t2, Wh2b, bh2, (float*)d_out, 256, 1024, 0, 0);
}
